// AtenMmQint8_54752243089862
// MI455X (gfx1250) — 8-bit matrix operand order repaired and checked by reading, compiled, not run on hardware
//
#include <hip/hip_runtime.h>
#include <stdint.h>


typedef int v8i __attribute__((ext_vector_type(8)));

#define M_DIM 8192
#define N_DIM 8192
#define K_DIM 128

__global__ __launch_bounds__(256) void quant_x_kernel(const float* __restrict__ x,
                                                      uint8_t* __restrict__ x8,
                                                      float* __restrict__ rx) {
    const int wave = threadIdx.x >> 5;
    const int lane = threadIdx.x & 31;
    const int row  = blockIdx.x * 8 + wave;

    const float4 f = *(const float4*)(x + (size_t)row * K_DIM + lane * 4);
    const uint32_t b0 = (uint32_t)(int)f.x & 255u;
    const uint32_t b1 = (uint32_t)(int)f.y & 255u;
    const uint32_t b2 = (uint32_t)(int)f.z & 255u;
    const uint32_t b3 = (uint32_t)(int)f.w & 255u;
    *(uint32_t*)(x8 + (size_t)row * K_DIM + lane * 4) =
        b0 | (b1 << 8) | (b2 << 16) | (b3 << 24);

    float s = f.x + f.y + f.z + f.w;
    #pragma unroll
    for (int off = 16; off > 0; off >>= 1) s += __shfl_xor(s, off, 32);
    if (lane == 0) rx[row] = s;
}

__global__ __launch_bounds__(256) void quant_y_kernel(const float* __restrict__ y,
                                                      uint8_t* __restrict__ y8t,
                                                      float* __restrict__ cy) {
    const int n = blockIdx.x * 256 + threadIdx.x;
    uint8_t* dst = y8t + (size_t)n * K_DIM;
    float s = 0.0f;
    #pragma unroll 4
    for (int k = 0; k < K_DIM; k += 4) {
        const float f0 = y[(size_t)(k + 0) * N_DIM + n];
        const float f1 = y[(size_t)(k + 1) * N_DIM + n];
        const float f2 = y[(size_t)(k + 2) * N_DIM + n];
        const float f3 = y[(size_t)(k + 3) * N_DIM + n];
        s += f0 + f1 + f2 + f3;
        const uint32_t b0 = (uint32_t)(int)f0 & 255u;
        const uint32_t b1 = (uint32_t)(int)f1 & 255u;
        const uint32_t b2 = (uint32_t)(int)f2 & 255u;
        const uint32_t b3 = (uint32_t)(int)f3 & 255u;
        *(uint32_t*)(dst + k) = b0 | (b1 << 8) | (b2 << 16) | (b3 << 24);
    }
    cy[n] = s;
}

__global__ __launch_bounds__(256) void gemm_iu8_kernel(const uint8_t* __restrict__ x8,
                                                       const uint8_t* __restrict__ y8t,
                                                       const float* __restrict__ rx,
                                                       const float* __restrict__ cy,
                                                       float* __restrict__ out) {
    const int wave   = threadIdx.x >> 5;
    const int lane   = threadIdx.x & 31;
    const int lane16 = lane & 15;
    const int half   = lane >> 4;

    const int wm = wave >> 2;
    const int wn = wave & 3;
    const int mBase = blockIdx.y * 128 + wm * 64;
    const int nBase = blockIdx.x * 256 + wn * 64;

    v8i acc[4][4];
    #pragma unroll
    for (int i = 0; i < 4; ++i)
        #pragma unroll
        for (int j = 0; j < 4; ++j) {
            v8i z = {};
            acc[i][j] = z;
        }

    #pragma unroll
    for (int kstep = 0; kstep < 2; ++kstep) {
        const int kbase = kstep * 64;

        v8i bfrag[4];
        #pragma unroll
        for (int ni = 0; ni < 4; ++ni) {
            const uint8_t* bp = y8t + (size_t)(nBase + ni * 16 + lane16) * K_DIM
                                    + kbase + half * 16;
            const uint4 q0 = *(const uint4*)(bp);
            const uint4 q1 = *(const uint4*)(bp + 32);
            v8i t;
            t[0] = (int)q0.x; t[1] = (int)q0.y; t[2] = (int)q0.z; t[3] = (int)q0.w;
            t[4] = (int)q1.x; t[5] = (int)q1.y; t[6] = (int)q1.z; t[7] = (int)q1.w;
            bfrag[ni] = t;
        }

        #pragma unroll
        for (int mi = 0; mi < 4; ++mi) {
            const uint8_t* ap = x8 + (size_t)(mBase + mi * 16 + lane16) * K_DIM
                                   + kbase + half * 16;
            const uint2 p0 = *(const uint2*)(ap);
            const uint2 p1 = *(const uint2*)(ap + 8);
            const uint2 p2 = *(const uint2*)(ap + 32);
            const uint2 p3 = *(const uint2*)(ap + 40);
            v8i a;
            a[0] = (int)p0.x; a[1] = (int)p0.y; a[2] = (int)p1.x; a[3] = (int)p1.y;
            a[4] = (int)p2.x; a[5] = (int)p2.y; a[6] = (int)p3.x; a[7] = (int)p3.y;

            #pragma unroll
            for (int ni = 0; ni < 4; ++ni) {
                acc[mi][ni] = __builtin_amdgcn_wmma_i32_16x16x64_iu8(
                    false, a, false, bfrag[ni], acc[mi][ni], false, false);
            }
        }
    }

    const float sxsy = 0.0215f * 0.0176f;
    float cyv[4];
    #pragma unroll
    for (int ni = 0; ni < 4; ++ni)
        cyv[ni] = 25.0f * cy[nBase + ni * 16 + lane16] - 57600.0f;

    #pragma unroll
    for (int mi = 0; mi < 4; ++mi) {
        float rxv[8];
        #pragma unroll
        for (int r = 0; r < 8; ++r)
            rxv[r] = 18.0f * rx[mBase + mi * 16 + half * 8 + r];

        #pragma unroll
        for (int ni = 0; ni < 4; ++ni) {
            const int col = nBase + ni * 16 + lane16;
            #pragma unroll
            for (int r = 0; r < 8; ++r) {
                const int row = mBase + mi * 16 + half * 8 + r;
                const float v = (float)acc[mi][ni][r];
                out[(size_t)row * N_DIM + col] = sxsy * (v - rxv[r] + cyv[ni]);
            }
        }
    }
}

extern "C" void kernel_launch(void* const* d_in, const int* in_sizes, int n_in,
                              void* d_out, int out_size, void* d_ws, size_t ws_size,
                              hipStream_t stream) {
    const float* x = (const float*)d_in[0];
    const float* y = (const float*)d_in[1];
    float* out = (float*)d_out;

    uint8_t* x8  = (uint8_t*)d_ws;
    uint8_t* y8t = x8 + (size_t)M_DIM * K_DIM;
    float*   rx  = (float*)(y8t + (size_t)N_DIM * K_DIM);
    float*   cy  = rx + M_DIM;

    quant_x_kernel<<<M_DIM / 8, 256, 0, stream>>>(x, x8, rx);
    quant_y_kernel<<<N_DIM / 256, 256, 0, stream>>>(y, y8t, cy);

    dim3 grid(N_DIM / 256, M_DIM / 128);
    gemm_iu8_kernel<<<grid, 256, 0, stream>>>(x8, y8t, rx, cy, out);
}
